// MultiHeadAttention_45079976739014
// MI455X (gfx1250) — compile-verified, matrix operand order checked by reading, not run on hardware
//
#include <hip/hip_runtime.h>

// ---------------------------------------------------------------------------
// MHA block for MI455X (gfx1250, wave32): bf16 WMMA GEMMs + flash attention
// + TDM tensor_load_to_lds, async global->LDS copies, global prefetch
// ---------------------------------------------------------------------------

typedef __bf16 bf16_t;
typedef bf16_t v16bf __attribute__((ext_vector_type(16)));
typedef float  v8f   __attribute__((ext_vector_type(8)));

typedef unsigned int uint32x4 __attribute__((ext_vector_type(4)));
typedef int          int32x4  __attribute__((ext_vector_type(4)));
typedef int          int32x8  __attribute__((ext_vector_type(8)));

#define DMODEL 1024
#define HEADS  16
#define DK     64
#define SEQ    2048
#define BATCH  4
#define MTOT   (BATCH * SEQ)   // 8192 rows

// fp32 -> bf16 round-to-nearest-even
__device__ __forceinline__ unsigned short f2bf(float x) {
    union { float f; unsigned int u; } v; v.f = x;
    unsigned int r = v.u + 0x7FFFu + ((v.u >> 16) & 1u);
    return (unsigned short)(r >> 16);
}

__device__ __forceinline__ v8f zero8() {
    v8f z;
    #pragma unroll
    for (int i = 0; i < 8; ++i) z[i] = 0.0f;
    return z;
}

// Load a 16x32 bf16 operand fragment (A-matrix layout per CDNA5 ISA 7.12.2)
// from a row-major LDS tile. Two 16B ds_load_b128 per lane.
__device__ __forceinline__ v16bf load_frag(const unsigned short* base,
                                           int rowBase, int rowStride,
                                           int kBase, int lane) {
    const int r  = lane & 15;
    const int kb = (lane >> 4) * 8;
    const unsigned short* p = base + (rowBase + r) * rowStride + kBase + kb;
    union { v16bf v; uint4 u[2]; } f;
    f.u[0] = *(const uint4*)(p);
    f.u[1] = *(const uint4*)(p + 16);
    return f.v;
}

__device__ __forceinline__ v8f wmma_bf16(v16bf a, v16bf b, v8f c) {
    return __builtin_amdgcn_wmma_f32_16x16x32_bf16(
        false, a, false, b, (short)0, c, false, false);
}

// Per-lane async copy of 16 bytes global -> LDS (tracked by ASYNCcnt).
// VDST = LDS byte offset VGPR, VADDR = 64-bit global address, saddr = off.
__device__ __forceinline__ void async_copy_b128(unsigned ldsByteOff,
                                                const void* gptr) {
    unsigned long long ga = (unsigned long long)(size_t)gptr;
    asm volatile("global_load_async_to_lds_b128 %0, %1, off"
                 :: "v"(ldsByteOff), "v"(ga) : "memory");
}

__device__ __forceinline__ void wait_asynccnt0() {
    asm volatile("s_wait_asynccnt 0x0" ::: "memory");
}

// TDM: DMA a rows x cols tile of bf16 from a row-major [*, DMODEL] matrix
// into LDS (contiguous row-major). Descriptor per CDNA5 ISA 8.3/8.4.
// Must be executed by a single wave (TDM ignores EXEC).
__device__ __forceinline__ void tdm_load_tile_bf16(unsigned ldsByteOff,
                                                   const void* gtile,
                                                   unsigned tileRows,
                                                   unsigned tileCols) {
    unsigned long long ga = (unsigned long long)(size_t)gtile;
    uint32x4 g0;
    g0[0] = 1u;                                          // count=1, user mode
    g0[1] = ldsByteOff;                                  // lds_addr
    g0[2] = (unsigned)(ga & 0xFFFFFFFFull);              // global_addr[31:0]
    g0[3] = (unsigned)((ga >> 32) & 0x01FFFFFFull)       // global_addr[56:32]
          | (2u << 30);                                  // type=2 (image)
    int32x8 g1;
    g1[0] = (int)(1u << 16);                             // data_size=1 (2B)
    g1[1] = (int)((DMODEL & 0xFFFFu) << 16);             // tensor_dim0 lo16
    g1[2] = (int)((DMODEL >> 16) | ((MTOT & 0xFFFFu) << 16)); // d0 hi / d1 lo
    g1[3] = (int)((MTOT >> 16) | (tileRows == 0 ? 0 : (tileCols << 16)));
    //        tensor_dim1 hi16            | tile_dim0
    g1[4] = (int)(tileRows);                             // tile_dim1, tile_dim2=0
    g1[5] = (int)(DMODEL);                               // tensor_dim0_stride lo32
    g1[6] = 0;                                           // stride hi / d1_stride lo
    g1[7] = 0;
    int32x4 gz;
    gz[0] = 0; gz[1] = 0; gz[2] = 0; gz[3] = 0;
#if __has_include(<hip/amd_detail/amd_gfx1250_TDM.h>)
    int32x8 gz8;
    #pragma unroll
    for (int i = 0; i < 8; ++i) gz8[i] = 0;
    __builtin_amdgcn_tensor_load_to_lds(g0, g1, gz, gz, gz8, 0);
#else
    __builtin_amdgcn_tensor_load_to_lds(g0, g1, gz, gz, 0);
#endif
}

// ---------------------------------------------------------------------------
// Kernel 1: P = X @ W^T + b   (fp32 in, bf16 out)
// grid = (MTOT/64, DMODEL/64), block = 128 (4 waves); 64x64 tile per block
// ---------------------------------------------------------------------------
__global__ __launch_bounds__(128) void proj_kernel(
    const float* __restrict__ X, const float* __restrict__ W,
    const float* __restrict__ bias, unsigned short* __restrict__ P) {

    __shared__ unsigned short As[64 * 32];
    __shared__ unsigned short Bs[64 * 32];

    const int t    = threadIdx.x;
    const int lane = t & 31;
    const int wave = t >> 5;
    const int m0   = blockIdx.x * 64;
    const int n0   = blockIdx.y * 64;

    v8f acc[4];
    #pragma unroll
    for (int i = 0; i < 4; ++i) acc[i] = zero8();

    for (int kk = 0; kk < DMODEL; kk += 32) {
        // prefetch next K-panel while this one is staged/consumed
        if (kk + 32 < DMODEL) {
            const int pr = t >> 1, pc = kk + 32 + (t & 1) * 16;
            __builtin_prefetch(&X[(size_t)(m0 + pr) * DMODEL + pc], 0, 1);
            __builtin_prefetch(&W[(size_t)(n0 + pr) * DMODEL + pc], 0, 1);
        }
        // cooperative fp32->bf16 staging: 64x32 tiles of X and of W rows
        #pragma unroll
        for (int i = 0; i < 4; ++i) {
            const int e   = t + i * 128;
            const int row = e >> 3;
            const int c4  = e & 7;
            const float4 xa = *(const float4*)&X[(size_t)(m0 + row) * DMODEL + kk + c4 * 4];
            unsigned short* da = &As[row * 32 + c4 * 4];
            da[0] = f2bf(xa.x); da[1] = f2bf(xa.y);
            da[2] = f2bf(xa.z); da[3] = f2bf(xa.w);
            const float4 wa = *(const float4*)&W[(size_t)(n0 + row) * DMODEL + kk + c4 * 4];
            unsigned short* db = &Bs[row * 32 + c4 * 4];
            db[0] = f2bf(wa.x); db[1] = f2bf(wa.y);
            db[2] = f2bf(wa.z); db[3] = f2bf(wa.w);
        }
        __syncthreads();

        const v16bf a = load_frag(As, wave * 16, 32, 0, lane);
        #pragma unroll
        for (int nt = 0; nt < 4; ++nt) {
            const v16bf b = load_frag(Bs, nt * 16, 32, 0, lane);
            acc[nt] = wmma_bf16(a, b, acc[nt]);
        }
        __syncthreads();
    }

    const int n    = lane & 15;
    const int mOff = (lane >> 4) * 8;
    #pragma unroll
    for (int nt = 0; nt < 4; ++nt) {
        const int col = n0 + nt * 16 + n;
        const float bv = bias[col];
        #pragma unroll
        for (int r = 0; r < 8; ++r) {
            const int row = m0 + wave * 16 + r + mOff;
            P[(size_t)row * DMODEL + col] = f2bf(acc[nt][r] + bv);
        }
    }
}

// ---------------------------------------------------------------------------
// Kernel 2: causal flash attention per (q-tile of 64, head, batch)
// block = 128 (4 waves); each wave owns 16 query rows
// Q tile: async global->LDS copies.  K tile: TDM tensor_load_to_lds.
// ---------------------------------------------------------------------------
__global__ __launch_bounds__(128) void flash_kernel(
    const unsigned short* __restrict__ Qp, const unsigned short* __restrict__ Kp,
    const unsigned short* __restrict__ Vp, float* __restrict__ Ctx) {

    __shared__ unsigned short Qs[64 * 64];      // [qrow][dk]
    __shared__ unsigned short Ks[64 * 64];      // [key][dk]   (B tile for QK^T)
    __shared__ unsigned short Vt[64 * 64];      // [dk][key]   (B tile for P*V)
    __shared__ unsigned short Ps[4][16 * 64];   // per-wave P  [qrow][key]

    const int t    = threadIdx.x;
    const int lane = t & 31;
    const int wave = t >> 5;
    const int qt   = blockIdx.x;
    const int h    = blockIdx.y;
    const int b    = blockIdx.z;

    const size_t rowBase = (size_t)b * SEQ;
    const int    colBase = h * DK;

    const unsigned QsBase = (unsigned)(size_t)(void*)Qs;
    const unsigned KsBase = (unsigned)(size_t)(void*)Ks;

    // stage Q tile once via async global->LDS (ASYNCcnt-tracked)
    #pragma unroll
    for (int i = 0; i < 4; ++i) {
        const int e = t + i * 128, row = e >> 3, c8 = e & 7;
        async_copy_b128(QsBase + (unsigned)(row * 64 + c8 * 8) * 2,
                        &Qp[(rowBase + qt * 64 + row) * DMODEL + colBase + c8 * 8]);
    }

    v8f o[4];
    float m_r[8], l_r[8];
    #pragma unroll
    for (int i = 0; i < 4; ++i) o[i] = zero8();
    #pragma unroll
    for (int r = 0; r < 8; ++r) { m_r[r] = -3.0e38f; l_r[r] = 0.0f; }

    const int n    = lane & 15;
    const int mOff = (lane >> 4) * 8;

    wait_asynccnt0();   // Q tile resident before first use

    for (int kt = 0; kt <= qt; ++kt) {
        __syncthreads();   // previous iteration's LDS reads done

        // K tile: single TDM descriptor, issued by wave 0 only
        if (wave == 0) {
            tdm_load_tile_bf16(KsBase,
                               &Kp[(rowBase + (size_t)kt * 64) * DMODEL + colBase],
                               64u, 64u);
        }
        // prefetch next K/V tiles into cache while this tile computes
        if (kt + 1 <= qt) {
            const size_t nrow = rowBase + (size_t)(kt + 1) * 64 + (t >> 1);
            __builtin_prefetch(&Kp[nrow * DMODEL + colBase + (t & 1) * 32], 0, 1);
            __builtin_prefetch(&Vp[nrow * DMODEL + colBase + (t & 1) * 32], 0, 1);
        }
        // V tile: transposed into LDS ([dk][key]) so P*V B-frag is contiguous
        {
            const int key = t >> 1;
            const int c0  = (t & 1) * 32;
            #pragma unroll
            for (int j = 0; j < 4; ++j) {
                uint4 pk = *(const uint4*)&Vp[(rowBase + kt * 64 + key) * DMODEL + colBase + c0 + j * 8];
                const unsigned short* e8 = (const unsigned short*)&pk;
                #pragma unroll
                for (int q2 = 0; q2 < 8; ++q2)
                    Vt[(c0 + j * 8 + q2) * 64 + key] = e8[q2];
            }
        }
        if (wave == 0) __builtin_amdgcn_s_wait_tensorcnt(0);
        __syncthreads();

        // scores: S = Q * K^T   (dk=64 -> two 16x16x32 WMMA steps per n-tile)
        v8f s[4];
        #pragma unroll
        for (int i = 0; i < 4; ++i) s[i] = zero8();
        #pragma unroll
        for (int kk = 0; kk < DK; kk += 32) {
            const v16bf a = load_frag(Qs, wave * 16, 64, kk, lane);
            #pragma unroll
            for (int nt = 0; nt < 4; ++nt) {
                const v16bf bb = load_frag(Ks, nt * 16, 64, kk, lane);
                s[nt] = wmma_bf16(a, bb, s[nt]);
            }
        }

        // causal mask (BEFORE scale, like the reference), online softmax
        #pragma unroll
        for (int r = 0; r < 8; ++r) {
            const int qrow = qt * 64 + wave * 16 + r + mOff;
            float rowmax = -3.0e38f;
            float sv[4];
            #pragma unroll
            for (int nt = 0; nt < 4; ++nt) {
                const int key = kt * 64 + nt * 16 + n;
                float x = s[nt][r];
                x = (key > qrow) ? -1.0e9f : x;
                x *= 0.125f;                        // 1/sqrt(dk)
                sv[nt] = x;
                rowmax = fmaxf(rowmax, x);
            }
            #pragma unroll
            for (int off = 8; off >= 1; off >>= 1)
                rowmax = fmaxf(rowmax, __shfl_xor(rowmax, off, 32));
            const float mn   = fmaxf(m_r[r], rowmax);
            const float corr = __expf(m_r[r] - mn);
            float rsum = 0.0f;
            #pragma unroll
            for (int nt = 0; nt < 4; ++nt) {
                const float p = __expf(sv[nt] - mn);
                rsum += p;
                s[nt][r] = p;
            }
            #pragma unroll
            for (int off = 8; off >= 1; off >>= 1)
                rsum += __shfl_xor(rsum, off, 32);
            l_r[r] = l_r[r] * corr + rsum;
            m_r[r] = mn;
            #pragma unroll
            for (int ot = 0; ot < 4; ++ot) o[ot][r] *= corr;
        }

        // re-layout P (C-format -> A-format) via per-wave LDS scratch
        #pragma unroll
        for (int r = 0; r < 8; ++r)
            #pragma unroll
            for (int nt = 0; nt < 4; ++nt)
                Ps[wave][(r + mOff) * 64 + nt * 16 + n] = f2bf(s[nt][r]);
        __syncthreads();

        // O += P * V   (K-dim = 64 keys -> two WMMA steps per dk n-tile)
        #pragma unroll
        for (int kk = 0; kk < 64; kk += 32) {
            const v16bf a = load_frag(Ps[wave], 0, 64, kk, lane);
            #pragma unroll
            for (int ot = 0; ot < 4; ++ot) {
                const v16bf bb = load_frag(Vt, ot * 16, 64, kk, lane);
                o[ot] = wmma_bf16(a, bb, o[ot]);
            }
        }
    }

    // normalize and write fp32 ctx
    #pragma unroll
    for (int r = 0; r < 8; ++r) {
        const float inv = 1.0f / l_r[r];
        const size_t row = rowBase + qt * 64 + wave * 16 + r + mOff;
        #pragma unroll
        for (int ot = 0; ot < 4; ++ot)
            Ctx[row * DMODEL + colBase + ot * 16 + n] = o[ot][r] * inv;
    }
}

// ---------------------------------------------------------------------------
// Kernel 3: out = LayerNorm(ctx + q) ; one block (256 thr) per row
// ---------------------------------------------------------------------------
__global__ __launch_bounds__(256) void ln_kernel(
    const float* __restrict__ Ctx, const float* __restrict__ Q,
    const float* __restrict__ gamma, const float* __restrict__ beta,
    float* __restrict__ Out) {

    const int row = blockIdx.x;
    const int t   = threadIdx.x;
    __shared__ float red[2][8];

    const float* cp = &Ctx[(size_t)row * DMODEL];
    const float* qp = &Q[(size_t)row * DMODEL];

    float x[4], sum = 0.0f, sq = 0.0f;
    #pragma unroll
    for (int i = 0; i < 4; ++i) {
        const int c = t + i * 256;
        x[i] = cp[c] + qp[c];
        sum += x[i];
        sq  += x[i] * x[i];
    }
    #pragma unroll
    for (int off = 16; off >= 1; off >>= 1) {
        sum += __shfl_xor(sum, off, 32);
        sq  += __shfl_xor(sq,  off, 32);
    }
    const int wave = t >> 5, lane = t & 31;
    if (lane == 0) { red[0][wave] = sum; red[1][wave] = sq; }
    __syncthreads();
    if (wave == 0) {
        float s  = (lane < 8) ? red[0][lane] : 0.0f;
        float q2 = (lane < 8) ? red[1][lane] : 0.0f;
        #pragma unroll
        for (int off = 4; off >= 1; off >>= 1) {
            s  += __shfl_xor(s,  off, 32);
            q2 += __shfl_xor(q2, off, 32);
        }
        if (lane == 0) { red[0][0] = s; red[1][0] = q2; }
    }
    __syncthreads();
    const float mu   = red[0][0] * (1.0f / DMODEL);
    const float var  = red[1][0] * (1.0f / DMODEL) - mu * mu;
    const float rstd = rsqrtf(var + 1e-5f);
    #pragma unroll
    for (int i = 0; i < 4; ++i) {
        const int c = t + i * 256;
        Out[(size_t)row * DMODEL + c] = (x[i] - mu) * rstd * gamma[c] + beta[c];
    }
}

// ---------------------------------------------------------------------------
extern "C" void kernel_launch(void* const* d_in, const int* in_sizes, int n_in,
                              void* d_out, int out_size, void* d_ws, size_t ws_size,
                              hipStream_t stream) {
    const float* q     = (const float*)d_in[0];
    const float* k     = (const float*)d_in[1];
    const float* v     = (const float*)d_in[2];
    // d_in[3] = trg_mask (int32, causal) -- applied analytically in-kernel
    const float* Wq    = (const float*)d_in[4];
    const float* bq    = (const float*)d_in[5];
    const float* Wk    = (const float*)d_in[6];
    const float* bk    = (const float*)d_in[7];
    const float* Wv    = (const float*)d_in[8];
    const float* bv    = (const float*)d_in[9];
    const float* gamma = (const float*)d_in[10];
    const float* beta  = (const float*)d_in[11];
    float* out = (float*)d_out;

    char* ws = (char*)d_ws;
    const size_t mat = (size_t)MTOT * DMODEL;
    unsigned short* Qp = (unsigned short*)(ws);
    unsigned short* Kp = (unsigned short*)(ws + mat * 2);
    unsigned short* Vp = (unsigned short*)(ws + 2 * mat * 2);
    float*          Cx = (float*)(ws + 3 * mat * 2);

    const dim3 gProj(MTOT / 64, DMODEL / 64, 1);
    proj_kernel<<<gProj, 128, 0, stream>>>(q, Wq, bq, Qp);
    proj_kernel<<<gProj, 128, 0, stream>>>(k, Wk, bk, Kp);
    proj_kernel<<<gProj, 128, 0, stream>>>(v, Wv, bv, Vp);

    flash_kernel<<<dim3(SEQ / 64, HEADS, BATCH), 128, 0, stream>>>(Qp, Kp, Vp, Cx);

    ln_kernel<<<MTOT, 256, 0, stream>>>(Cx, q, gamma, beta, out);
}
